// GraphAttention_21947282882895
// MI455X (gfx1250) — matrix operand order repaired and checked by reading, not compiled, not run on hardware
//
#include <hip/hip_runtime.h>
#include <hip/hip_bf16.h>
#include <stdint.h>

#define N_NODES 8192
#define F_IN    1024
#define F_OUT   64
#define H_HEADS 4
#define LEAKY   0.2f
#define MASKV   -10000000000.0f

typedef __attribute__((ext_vector_type(2)))  float  v2f;
typedef __attribute__((ext_vector_type(4)))  float  v4f;
typedef __attribute__((ext_vector_type(8)))  float  v8f;
typedef __attribute__((ext_vector_type(16))) __bf16 v16bf;

__device__ __forceinline__ float bperm_f(int srcLane, float v) {
  return __int_as_float(__builtin_amdgcn_ds_bpermute(srcLane << 2, __float_as_int(v)));
}
__device__ __forceinline__ float xor_f(float v, int mask) {
  int l = ((int)(threadIdx.x & 31)) ^ mask;
  return bperm_f(l, v);
}
__device__ __forceinline__ __bf16 f2bf(float f) {
  union { float f; uint32_t u; } c; c.f = f;
  uint32_t r = (c.u + 0x7FFFu + ((c.u >> 16) & 1u)) >> 16;
  uint16_t s = (uint16_t)r;
  __bf16 o; __builtin_memcpy(&o, &s, 2);
  return o;
}

// ---------------------------------------------------------------------------
// Kernel 0: transpose W (H,F,F_) -> Wt (H,F_,F) so the f32 WMMA B operand
// reads contiguously along K.
// ---------------------------------------------------------------------------
__global__ void wt_transpose_kernel(const float* __restrict__ W,
                                    float* __restrict__ Wt) {
  int idx = blockIdx.x * 256 + threadIdx.x;
  if (idx >= H_HEADS * F_IN * F_OUT) return;
  int k = idx & (F_OUT - 1);
  int f = (idx >> 6) & (F_IN - 1);
  int h = idx >> 16;
  Wt[((h * F_OUT + k) << 10) + f] = W[idx];
}

// ---------------------------------------------------------------------------
// Kernel 1: feats = X @ W[h] in fp32 via V_WMMA_F32_16X16X4_F32.
// One block = 16 rows of X, 4 waves = 4 heads. Each wave produces the
// 16x64 feats tile in 4 v8f accumulators, derives s_self/s_neigh by
// dotting with attn_a (cross-lane bpermute reduction), and writes the
// tile transposed as bf16 -> featsT[h][k][n] for kernel 2's B operand.
// ---------------------------------------------------------------------------
__global__ void __launch_bounds__(128)
feats_kernel(const float* __restrict__ X, const float* __restrict__ Wt,
             const float* __restrict__ attn_a,
             float* __restrict__ s_self, float* __restrict__ s_neigh,
             __bf16* __restrict__ featsT) {
  const int i0   = blockIdx.x * 16;
  const int h    = threadIdx.x >> 5;
  const int lane = threadIdx.x & 31;
  const int n    = lane & 15;
  const int half = lane >> 4;

  v8f acc0 = {0,0,0,0,0,0,0,0};
  v8f acc1 = {0,0,0,0,0,0,0,0};
  v8f acc2 = {0,0,0,0,0,0,0,0};
  v8f acc3 = {0,0,0,0,0,0,0,0};

  const float* Xrow = X + (size_t)(i0 + n) * F_IN;
  const float* W0 = Wt + ((size_t)h * F_OUT +  0 + n) * F_IN;
  const float* W1 = Wt + ((size_t)h * F_OUT + 16 + n) * F_IN;
  const float* W2 = Wt + ((size_t)h * F_OUT + 32 + n) * F_IN;
  const float* W3 = Wt + ((size_t)h * F_OUT + 48 + n) * F_IN;

  for (int k0 = 0; k0 < F_IN; k0 += 4) {
    const int kk = k0 + half * 2;           // 32-bit A/B layout: half picks K pair
    v2f a  = *(const v2f*)(Xrow + kk);
    v2f b0 = *(const v2f*)(W0 + kk);
    v2f b1 = *(const v2f*)(W1 + kk);
    v2f b2 = *(const v2f*)(W2 + kk);
    v2f b3 = *(const v2f*)(W3 + kk);
    acc0 = __builtin_amdgcn_wmma_f32_16x16x4_f32(false, a, false, b0, (short)0, acc0, false, false);
    acc1 = __builtin_amdgcn_wmma_f32_16x16x4_f32(false, a, false, b1, (short)0, acc1, false, false);
    acc2 = __builtin_amdgcn_wmma_f32_16x16x4_f32(false, a, false, b2, (short)0, acc2, false, false);
    acc3 = __builtin_amdgcn_wmma_f32_16x16x4_f32(false, a, false, b3, (short)0, acc3, false, false);
  }

  // attention logit projections: s = feats . a_self / a_neigh
  const float as0 = attn_a[h * 2 * F_OUT +  0 + n];
  const float as1 = attn_a[h * 2 * F_OUT + 16 + n];
  const float as2 = attn_a[h * 2 * F_OUT + 32 + n];
  const float as3 = attn_a[h * 2 * F_OUT + 48 + n];
  const float an0 = attn_a[h * 2 * F_OUT + F_OUT +  0 + n];
  const float an1 = attn_a[h * 2 * F_OUT + F_OUT + 16 + n];
  const float an2 = attn_a[h * 2 * F_OUT + F_OUT + 32 + n];
  const float an3 = attn_a[h * 2 * F_OUT + F_OUT + 48 + n];

  float ps[8], pn[8];
#pragma unroll
  for (int r = 0; r < 8; ++r) {
    ps[r] = acc0[r]*as0 + acc1[r]*as1 + acc2[r]*as2 + acc3[r]*as3;
    pn[r] = acc0[r]*an0 + acc1[r]*an1 + acc2[r]*an2 + acc3[r]*an3;
  }
#pragma unroll
  for (int d = 1; d < 16; d <<= 1) {
#pragma unroll
    for (int r = 0; r < 8; ++r) {
      ps[r] += xor_f(ps[r], d);
      pn[r] += xor_f(pn[r], d);
    }
  }
  if (n == 0) {
#pragma unroll
    for (int r = 0; r < 8; ++r) {
      s_self [h * N_NODES + i0 + half * 8 + r] = ps[r];
      s_neigh[h * N_NODES + i0 + half * 8 + r] = pn[r];
    }
  }

  // bf16 transposed feats: featsT[(h*64 + k)][node]; C layout: M = r + 8*half
#pragma unroll
  for (int r = 0; r < 8; ++r) {
    const int node = i0 + half * 8 + r;
    featsT[((size_t)(h * F_OUT +  0 + n)) * N_NODES + node] = f2bf(acc0[r]);
    featsT[((size_t)(h * F_OUT + 16 + n)) * N_NODES + node] = f2bf(acc1[r]);
    featsT[((size_t)(h * F_OUT + 32 + n)) * N_NODES + node] = f2bf(acc2[r]);
    featsT[((size_t)(h * F_OUT + 48 + n)) * N_NODES + node] = f2bf(acc3[r]);
  }
}

// ---------------------------------------------------------------------------
// Kernel 2: flash-style masked softmax + P@feats with v_wmma_f32_16x16x32_bf16.
// Block = 16 output rows, 4 waves = 4 heads (share A rows through cache).
// s_neigh staged through LDS with GLOBAL_LOAD_ASYNC_TO_LDS_B128 (ASYNCcnt),
// A stream prefetched ahead with global_prefetch_b8. P tile (16x32) built
// per-lane in contiguous K order (K = 16*half + e); feats block loaded as four
// 32x16 bf16 B operands from featsT. Online softmax with per-row alpha/l
// broadcast via ds_bpermute.
// ---------------------------------------------------------------------------
__global__ void __launch_bounds__(128)
gat_flash_kernel(const float* __restrict__ A,
                 const float* __restrict__ s_self,
                 const float* __restrict__ s_neigh,
                 const __bf16* __restrict__ featsT,
                 const float* __restrict__ bias,
                 float* __restrict__ out) {
  __shared__ __align__(16) float sn_sh[H_HEADS][1024];

  const int i0   = blockIdx.x * 16;
  const int h    = threadIdx.x >> 5;
  const int lane = threadIdx.x & 31;
  const int n    = lane & 15;
  const int half = lane >> 4;
  const int klo  = half * 8;      // row offset of this lane's accumulator elements
  const int row  = i0 + n;        // P-tile row handled by this lane

  const float sself = s_self[h * N_NODES + row];

  v8f acc0 = {0,0,0,0,0,0,0,0};
  v8f acc1 = {0,0,0,0,0,0,0,0};
  v8f acc2 = {0,0,0,0,0,0,0,0};
  v8f acc3 = {0,0,0,0,0,0,0,0};
  float m = -3.0e38f, lsum = 0.0f;

  const float*  Arow = A + (size_t)row * N_NODES;
  const __bf16* fb0 = featsT + ((size_t)(h * F_OUT +  0 + n)) * N_NODES + half * 16;
  const __bf16* fb1 = featsT + ((size_t)(h * F_OUT + 16 + n)) * N_NODES + half * 16;
  const __bf16* fb2 = featsT + ((size_t)(h * F_OUT + 32 + n)) * N_NODES + half * 16;
  const __bf16* fb3 = featsT + ((size_t)(h * F_OUT + 48 + n)) * N_NODES + half * 16;

  // low 32 bits of the generic pointer to LDS = wave-relative LDS byte address
  const uint32_t ldsBase = (uint32_t)(uintptr_t)(&sn_sh[0][0]);

  for (int jb = 0; jb < N_NODES; jb += 1024) {
    __syncthreads();
    {
      // async DMA: 4 heads x 1024 floats = 16KB, 8 x b128 per thread,
      // straight into LDS (no VGPR round-trip), tracked by ASYNCcnt.
#pragma unroll
      for (int u = 0; u < 8; ++u) {
        int s  = (int)threadIdx.x + u * 128;   // 0..1023 16B slots
        int h2 = s >> 8;
        int q  = s & 255;
        uint32_t ldsOff = ldsBase + (uint32_t)s * 16u;
        uint32_t goff   = (uint32_t)(h2 * (N_NODES * 4) + jb * 4 + q * 16);
        asm volatile("global_load_async_to_lds_b128 %0, %1, %2"
                     :
                     : "v"(ldsOff), "v"(goff), "s"(s_neigh)
                     : "memory");
      }
      asm volatile("s_wait_asynccnt 0" ::: "memory");
    }
    __syncthreads();

    for (int j0 = jb; j0 < jb + 1024; j0 += 32) {
      // keep the sequential A stream 8KB ahead (wraps within the row)
      __builtin_prefetch(Arow + ((j0 + 2048) & (N_NODES - 1)), 0, 3);

      float av[16], snv[16];
      {
        const v4f* ap = (const v4f*)(Arow + j0 + half * 16);
        v4f a0 = ap[0], a1 = ap[1], a2 = ap[2], a3 = ap[3];   // K: 16*half..16*half+15
        const v4f* sp = (const v4f*)(&sn_sh[h][(j0 - jb) + half * 16]);
        v4f s0 = sp[0], s1 = sp[1], s2 = sp[2], s3 = sp[3];
#pragma unroll
        for (int t = 0; t < 4; ++t) {
          av[t]      = a0[t]; av[4 + t]  = a1[t];
          av[8 + t]  = a2[t]; av[12 + t] = a3[t];
          snv[t]     = s0[t]; snv[4 + t] = s1[t];
          snv[8 + t] = s2[t]; snv[12 + t]= s3[t];
        }
      }

      float p[16];
      float bm = -3.0e38f;
#pragma unroll
      for (int t = 0; t < 16; ++t) {
        float s  = sself + snv[t];
        float lr = fmaxf(s, LEAKY * s);                 // leaky relu
        float sc = fmaf(MASKV, 1.0f - av[t], lr);       // mask
        p[t] = sc;
        bm = fmaxf(bm, sc);
      }
      bm = fmaxf(bm, xor_f(bm, 16));                    // combine K-halves of the row
      const float mnew  = fmaxf(m, bm);
      const float alpha = __expf(m - mnew);
      float rs = 0.0f;
#pragma unroll
      for (int t = 0; t < 16; ++t) { p[t] = __expf(p[t] - mnew); rs += p[t]; }
      rs += xor_f(rs, 16);
      lsum = lsum * alpha + rs;
      m = mnew;

      // rescale accumulators: row of acc element r is (klo + r)
#pragma unroll
      for (int r = 0; r < 8; ++r) {
        float avr = bperm_f(klo + r, alpha);
        acc0[r] *= avr; acc1[r] *= avr; acc2[r] *= avr; acc3[r] *= avr;
      }

      // pack P as the A operand, contiguous K (element e <-> K 16*half + e)
      v16bf pa;
#pragma unroll
      for (int t = 0; t < 16; ++t) pa[t] = f2bf(p[t]);

      v16bf b0 = *(const v16bf*)(fb0 + j0);
      v16bf b1 = *(const v16bf*)(fb1 + j0);
      v16bf b2 = *(const v16bf*)(fb2 + j0);
      v16bf b3 = *(const v16bf*)(fb3 + j0);
      acc0 = __builtin_amdgcn_wmma_f32_16x16x32_bf16(false, pa, false, b0, (short)0, acc0, false, false);
      acc1 = __builtin_amdgcn_wmma_f32_16x16x32_bf16(false, pa, false, b1, (short)0, acc1, false, false);
      acc2 = __builtin_amdgcn_wmma_f32_16x16x32_bf16(false, pa, false, b2, (short)0, acc2, false, false);
      acc3 = __builtin_amdgcn_wmma_f32_16x16x32_bf16(false, pa, false, b3, (short)0, acc3, false, false);
    }
  }

  // epilogue: normalize, bias, relu, store out[(node), h*64 + col]
#pragma unroll
  for (int r = 0; r < 8; ++r) {
    const float li  = bperm_f(klo + r, lsum);
    const float inv = 1.0f / li;
    const int node  = i0 + klo + r;
    float* orow = out + (size_t)node * (H_HEADS * F_OUT) + h * F_OUT;
    orow[ 0 + n] = fmaxf(fmaf(acc0[r], inv, bias[h * F_OUT +  0 + n]), 0.0f);
    orow[16 + n] = fmaxf(fmaf(acc1[r], inv, bias[h * F_OUT + 16 + n]), 0.0f);
    orow[32 + n] = fmaxf(fmaf(acc2[r], inv, bias[h * F_OUT + 32 + n]), 0.0f);
    orow[48 + n] = fmaxf(fmaf(acc3[r], inv, bias[h * F_OUT + 48 + n]), 0.0f);
  }
}

// ---------------------------------------------------------------------------
extern "C" void kernel_launch(void* const* d_in, const int* in_sizes, int n_in,
                              void* d_out, int out_size, void* d_ws, size_t ws_size,
                              hipStream_t stream) {
  (void)in_sizes; (void)n_in; (void)out_size; (void)ws_size;

  const float* X      = (const float*)d_in[0];
  const float* A      = (const float*)d_in[1];
  const float* W      = (const float*)d_in[2];
  const float* attn_a = (const float*)d_in[3];
  const float* bias   = (const float*)d_in[4];
  float* out = (float*)d_out;

  char* ws = (char*)d_ws;
  float*  Wt      = (float*)ws;                                     // 1 MB
  float*  s_self  = (float*)(ws + (1u << 20));                      // 128 KB
  float*  s_neigh = (float*)(ws + (1u << 20) + (128u << 10));       // 128 KB
  __bf16* featsT  = (__bf16*)(ws + (1u << 20) + (256u << 10));      // 4 MB

  wt_transpose_kernel<<<(H_HEADS * F_IN * F_OUT) / 256, 256, 0, stream>>>(W, Wt);
  feats_kernel<<<N_NODES / 16, 128, 0, stream>>>(X, Wt, attn_a, s_self, s_neigh, featsT);
  gat_flash_kernel<<<N_NODES / 16, 128, 0, stream>>>(A, s_self, s_neigh, featsT, bias, out);
}
